// Transformer_50483045597631
// MI455X (gfx1250) — matrix operand order repaired and checked by reading, not compiled, not run on hardware
//
#include <hip/hip_runtime.h>

// ---------------------------------------------------------------------------
// DAB-DETR decoder layer forward for MI455X (gfx1250, wave32, WMMA bf16).
// Branchless fragment loads (clamped addresses, b128 vector loads), zero-padded
// ragged-K attention buffers, f32 accumulation.
// ---------------------------------------------------------------------------
#define NQd   300
#define BBd   32
#define Dd    256
#define NHd   8
#define HWd   1024
#define DFFd  2048
#define SPAD  320   // NQd padded to a multiple of 32 for attn-prob K dim

typedef unsigned short u16b;  // raw bf16 bits
typedef __attribute__((ext_vector_type(16))) __bf16    v16bf;
typedef __attribute__((ext_vector_type(8)))  float     v8f;
typedef __attribute__((ext_vector_type(4)))  unsigned  u32x4;

union FragBF { unsigned u[8]; u32x4 q[2]; v16bf v; };
union FragF  { v8f v; float f[8]; };

__device__ __forceinline__ u16b f2bf(float f) {
    unsigned u = __builtin_bit_cast(unsigned, f);
    unsigned r = u + 0x7FFFu + ((u >> 16) & 1u);   // round-to-nearest-even
    return (u16b)(r >> 16);
}
__device__ __forceinline__ int imin(int a, int b) { return a < b ? a : b; }

// ---------------------------------------------------------------------------
__global__ void cvt_bf16_kernel(const float* __restrict__ in,
                                u16b* __restrict__ out, long n) {
    long i = (long)blockIdx.x * blockDim.x + threadIdx.x;
    if (i < n) out[i] = f2bf(in[i]);
}

// ---------------------------------------------------------------------------
// Generic batched WMMA GEMM.
//   BT      : true  -> C = A[M,K] * B[N,K]^T   (linear layers, Q·K^T)
//             false -> C = A[M,K] * B[K,N]     (attn · V)
//   OUTMODE : 0 = f32 out, 1 = bf16 out, 2 = bf16 out + relu
// Dual-pair: if A1 != null, accumulates A0*B0 + A1*B1 (same Kloop / strides).
// Kloop: loop K bound, multiple of 32 (A must be zero-padded to Kloop if
//        ragged). Kreal: valid B rows for the NN path (k clamped, not zeroed).
// A-row / B-col indices are CLAMPED, not guarded: out-of-range rows/cols only
// affect C entries that the epilogue drops, so the hot loop is branchless.
// Block = 128 threads (4 waves); wave tile 32x32; block tile 64x64.
// ---------------------------------------------------------------------------
template<bool BT, int OUTMODE>
__global__ void gemm_wmma_kernel(
    const u16b* __restrict__ A0p, const u16b* __restrict__ A1p,
    const u16b* __restrict__ B0p, const u16b* __restrict__ B1p,
    const float* __restrict__ bias0, const float* __restrict__ bias1,
    void* __restrict__ Cout,
    int M, int N, int Kloop, int Kreal, int lda, int ldb, int ldc,
    long aOffB, long aOffH, long bOffB, long bOffH, long cOffB, long cOffH,
    int nh)
{
    const int z  = blockIdx.z;
    const int bi = z / nh, hi = z - bi * nh;
    const long aoff = (long)bi * aOffB + (long)hi * aOffH;
    const long boff = (long)bi * bOffB + (long)hi * bOffH;
    const long coff = (long)bi * cOffB + (long)hi * cOffH;

    const int wave = threadIdx.x >> 5;
    const int lane = threadIdx.x & 31;
    const int lm = lane & 15, lh = lane >> 4;
    const int mBase = blockIdx.y * 64 + (wave >> 1) * 32;
    const int nBase = blockIdx.x * 64 + (wave & 1) * 32;

    // clamped per-lane row/col indices (branchless edge handling)
    const int mA0 = imin(mBase + lm,      M - 1);
    const int mA1 = imin(mBase + 16 + lm, M - 1);
    const int nB0 = imin(nBase + lm,      N - 1);
    const int nB1 = imin(nBase + 16 + lm, N - 1);

    v8f acc[2][2] = {};

    for (int p = 0; p < 2; ++p) {
        const u16b* Ap = p ? A1p : A0p;
        const u16b* Bp = p ? B1p : B0p;
        if (Ap == nullptr) continue;
        const u16b* A = Ap + aoff;
        const u16b* B = Bp + boff;

        // hoisted fragment base pointers
        const u16b* pa0 = A + (long)mA0 * lda + lh * 16;
        const u16b* pa1 = A + (long)mA1 * lda + lh * 16;
        const u16b* pb0;
        const u16b* pb1;
        if constexpr (BT) {
            pb0 = B + (long)nB0 * ldb + lh * 16;
            pb1 = B + (long)nB1 * ldb + lh * 16;
        } else {
            pb0 = B + nB0;
            pb1 = B + nB1;
        }

        for (int kb = 0; kb < Kloop; kb += 32) {
            FragBF a0, a1, b0, b1;
            // ---- A fragments: 2 x b128 each (contiguous along K) ----
            a0.q[0] = *(const u32x4*)(pa0 + kb);
            a0.q[1] = *(const u32x4*)(pa0 + kb + 8);
            a1.q[0] = *(const u32x4*)(pa1 + kb);
            a1.q[1] = *(const u32x4*)(pa1 + kb + 8);
            // ---- B fragments ----
            if constexpr (BT) {
                b0.q[0] = *(const u32x4*)(pb0 + kb);
                b0.q[1] = *(const u32x4*)(pb0 + kb + 8);
                b1.q[0] = *(const u32x4*)(pb1 + kb);
                b1.q[1] = *(const u32x4*)(pb1 + kb + 8);
            } else {
                #pragma unroll
                for (int v = 0; v < 8; ++v) {
                    const int kraw = kb + lh * 16 + v * 2;
                    const long k0 = imin(kraw,     Kreal - 1);
                    const long k1 = imin(kraw + 1, Kreal - 1);
                    unsigned lo0 = pb0[k0 * ldb], hi0 = pb0[k1 * ldb];
                    unsigned lo1 = pb1[k0 * ldb], hi1 = pb1[k1 * ldb];
                    b0.u[v] = lo0 | (hi0 << 16);
                    b1.u[v] = lo1 | (hi1 << 16);
                }
            }
            acc[0][0] = __builtin_amdgcn_wmma_f32_16x16x32_bf16(
                false, a0.v, false, b0.v, (short)0, acc[0][0], false, false);
            acc[0][1] = __builtin_amdgcn_wmma_f32_16x16x32_bf16(
                false, a0.v, false, b1.v, (short)0, acc[0][1], false, false);
            acc[1][0] = __builtin_amdgcn_wmma_f32_16x16x32_bf16(
                false, a1.v, false, b0.v, (short)0, acc[1][0], false, false);
            acc[1][1] = __builtin_amdgcn_wmma_f32_16x16x32_bf16(
                false, a1.v, false, b1.v, (short)0, acc[1][1], false, false);
        }
    }

    // ---- epilogue: bias, (relu), guarded store (runs once) ----
    #pragma unroll
    for (int i = 0; i < 2; ++i) {
        #pragma unroll
        for (int j = 0; j < 2; ++j) {
            const int n = nBase + j * 16 + lm;
            if (n >= N) continue;
            float bs = 0.f;
            if (bias0) bs += bias0[n];
            if (bias1) bs += bias1[n];
            FragF cf; cf.v = acc[i][j];
            #pragma unroll
            for (int r = 0; r < 8; ++r) {
                const int m = mBase + i * 16 + r + lh * 8;
                if (m >= M) continue;
                float val = cf.f[r] + bs;
                if (OUTMODE == 2) val = val > 0.f ? val : 0.f;
                const long idx = coff + (long)m * ldc + n;
                if (OUTMODE == 0) ((float*)Cout)[idx] = val;
                else              ((u16b*)Cout)[idx]  = f2bf(val);
            }
        }
    }
}

// ---------------------------------------------------------------------------
// Row softmax with scale: f32 in -> bf16 out; zero-fills [cols, colsPad).
// One 256-thread block per row; both buffers have row stride `ld`.
// ---------------------------------------------------------------------------
__global__ void softmax_kernel(const float* __restrict__ S,
                               u16b* __restrict__ O,
                               int cols, int colsPad, int ld, float scale) {
    const long row = blockIdx.x;
    const float* x = S + row * (long)ld;
    u16b* y = O + row * (long)ld;
    __shared__ float red[256];
    const int t = threadIdx.x;

    float mx = -3.4e38f;
    for (int i = t; i < cols; i += 256) mx = fmaxf(mx, x[i] * scale);
    red[t] = mx; __syncthreads();
    for (int s = 128; s > 0; s >>= 1) {
        if (t < s) red[t] = fmaxf(red[t], red[t + s]);
        __syncthreads();
    }
    mx = red[0]; __syncthreads();

    float sum = 0.f;
    for (int i = t; i < cols; i += 256) sum += __expf(x[i] * scale - mx);
    red[t] = sum; __syncthreads();
    for (int s = 128; s > 0; s >>= 1) {
        if (t < s) red[t] += red[t + s];
        __syncthreads();
    }
    const float inv = 1.0f / red[0];
    for (int i = t; i < colsPad; i += 256) {
        float e = (i < cols) ? __expf(x[i] * scale - mx) * inv : 0.f;
        y[i] = f2bf(e);
    }
}

// ---------------------------------------------------------------------------
// out = LayerNorm(X + Y) * g + b ; writes f32 (outF) and/or bf16 (outH).
// One 256-thread block per row, D == 256.
// ---------------------------------------------------------------------------
__global__ void add_ln_kernel(const float* __restrict__ X,
                              const float* __restrict__ Y,
                              const float* __restrict__ g,
                              const float* __restrict__ b,
                              float* __restrict__ outF,
                              u16b* __restrict__ outH) {
    const int t = threadIdx.x;
    const long base = (long)blockIdx.x * Dd;
    float v = X[base + t] + Y[base + t];
    __shared__ float red[Dd];
    red[t] = v; __syncthreads();
    for (int s = Dd / 2; s > 0; s >>= 1) {
        if (t < s) red[t] += red[t + s];
        __syncthreads();
    }
    const float mu = red[0] * (1.0f / Dd); __syncthreads();
    const float d = v - mu;
    red[t] = d * d; __syncthreads();
    for (int s = Dd / 2; s > 0; s >>= 1) {
        if (t < s) red[t] += red[t + s];
        __syncthreads();
    }
    const float var = red[0] * (1.0f / Dd);
    const float o = d * rsqrtf(var + 1e-5f) * g[t] + b[t];
    if (outF) outF[base + t] = o;
    if (outH) outH[base + t] = f2bf(o);
}

// ---------------------------------------------------------------------------
// host-side launch helpers
// ---------------------------------------------------------------------------
static void cvt(hipStream_t s, const float* in, u16b* out, long n) {
    hipLaunchKernelGGL(cvt_bf16_kernel, dim3((unsigned)((n + 255) / 256)),
                       dim3(256), 0, s, in, out, n);
}

static void gemm(hipStream_t s, bool bt, int outmode,
                 const u16b* A0, const u16b* A1,
                 const u16b* B0, const u16b* B1,
                 const float* b0, const float* b1, void* C,
                 int M, int N, int Kloop, int Kreal,
                 int lda, int ldb, int ldc,
                 long aOffB = 0, long aOffH = 0, long bOffB = 0, long bOffH = 0,
                 long cOffB = 0, long cOffH = 0, int nh = 1, int Z = 1) {
    dim3 grid((N + 63) / 64, (M + 63) / 64, Z), blk(128);
#define GEMM_ARGS A0, A1, B0, B1, b0, b1, C, M, N, Kloop, Kreal, \
                  lda, ldb, ldc, aOffB, aOffH, bOffB, bOffH, cOffB, cOffH, nh
    if (bt) {
        if (outmode == 0)
            hipLaunchKernelGGL((gemm_wmma_kernel<true, 0>), grid, blk, 0, s, GEMM_ARGS);
        else if (outmode == 1)
            hipLaunchKernelGGL((gemm_wmma_kernel<true, 1>), grid, blk, 0, s, GEMM_ARGS);
        else
            hipLaunchKernelGGL((gemm_wmma_kernel<true, 2>), grid, blk, 0, s, GEMM_ARGS);
    } else {
        hipLaunchKernelGGL((gemm_wmma_kernel<false, 1>), grid, blk, 0, s, GEMM_ARGS);
    }
#undef GEMM_ARGS
}

extern "C" void kernel_launch(void* const* d_in, const int* in_sizes, int n_in,
                              void* d_out, int out_size, void* d_ws, size_t ws_size,
                              hipStream_t stream) {
    (void)in_sizes; (void)n_in; (void)out_size; (void)ws_size;
    const float* tgt    = (const float*)d_in[0];
    const float* memory = (const float*)d_in[1];
    const float* pos    = (const float*)d_in[2];
    const float* qpos   = (const float*)d_in[3];
    const float* qse    = (const float*)d_in[4];
    const float* sa_w   = (const float*)d_in[5];
    const float* sa_b   = (const float*)d_in[6];
    const float* ca_w   = (const float*)d_in[7];
    const float* ca_b   = (const float*)d_in[8];
    const float* out_w  = (const float*)d_in[9];
    const float* out_b  = (const float*)d_in[10];
    const float* ffn_w1 = (const float*)d_in[11];
    const float* ffn_b1 = (const float*)d_in[12];
    const float* ffn_w2 = (const float*)d_in[13];
    const float* ffn_b2 = (const float*)d_in[14];
    const float* ln_g   = (const float*)d_in[15];
    const float* ln_b   = (const float*)d_in[16];

    const long MQ = (long)NQd * BBd;   // 9600 query rows
    const long MM = (long)HWd * BBd;   // 32768 memory rows
    const long WW = (long)Dd * Dd;     // 65536 per square weight
    const int  LD = BBd * Dd;          // 8192 row stride of [L,B,D]
    const int  hd = 32;                // per-head content dim

    char* w = (char*)d_ws; size_t off = 0;
    auto alloc = [&](size_t bytes) -> void* {
        void* p = w + off; off = (off + bytes + 255) & ~(size_t)255; return p;
    };
    auto allocH = [&](long n) { return (u16b*)alloc((size_t)n * 2); };
    auto allocF = [&](long n) { return (float*)alloc((size_t)n * 4); };

    // bf16 input mirrors
    u16b* tgt16  = allocH(MQ * Dd);
    u16b* qpos16 = allocH(MQ * Dd);
    u16b* qse16  = allocH(MQ * Dd);
    u16b* mem16  = allocH(MM * Dd);
    u16b* pos16  = allocH(MM * Dd);
    u16b* wsa16  = allocH(5 * WW);
    u16b* wca16  = allocH(6 * WW);
    u16b* wout16 = allocH(2 * WW);
    u16b* wf116  = allocH((long)DFFd * Dd);
    u16b* wf216  = allocH((long)Dd * DFFd);
    // self-attention (score rows padded NQd -> SPAD in the column dim)
    u16b*  q16   = allocH(MQ * Dd);
    u16b*  k16   = allocH(MQ * Dd);
    u16b*  v16   = allocH(MQ * Dd);
    float* scS   = allocF((long)BBd * NHd * NQd * SPAD);
    u16b*  atS16 = allocH((long)BBd * NHd * NQd * SPAD);
    u16b*  ao16  = allocH(MQ * Dd);
    float* aoF   = allocF(MQ * Dd);
    float* t1F   = allocF(MQ * Dd);
    u16b*  t1_16 = allocH(MQ * Dd);
    // cross-attention
    u16b*  qsum16 = allocH(MQ * Dd);
    u16b*  qsep16 = allocH(MQ * Dd);
    u16b*  ksum16 = allocH(MM * Dd);
    u16b*  kpos16 = allocH(MM * Dd);
    u16b*  vmem16 = allocH(MM * Dd);
    float* scC    = allocF((long)BBd * NHd * NQd * HWd);
    u16b*  atC16  = allocH((long)BBd * NHd * NQd * HWd);
    u16b*  ao2_16 = allocH(MQ * Dd);
    float* ao2F   = allocF(MQ * Dd);
    float* t2F    = allocF(MQ * Dd);
    u16b*  t2_16  = allocH(MQ * Dd);
    // FFN
    u16b*  h16    = allocH(MQ * DFFd);
    float* ffnF   = allocF(MQ * Dd);
    // slack so clamped-row overreads near the end of the arena stay in-bounds
    (void)alloc(1 << 20);

    // ---- 0) f32 -> bf16 conversions ----
    cvt(stream, tgt,    tgt16,  MQ * Dd);
    cvt(stream, qpos,   qpos16, MQ * Dd);
    cvt(stream, qse,    qse16,  MQ * Dd);
    cvt(stream, memory, mem16,  MM * Dd);
    cvt(stream, pos,    pos16,  MM * Dd);
    cvt(stream, sa_w,   wsa16,  5 * WW);
    cvt(stream, ca_w,   wca16,  6 * WW);
    cvt(stream, out_w,  wout16, 2 * WW);
    cvt(stream, ffn_w1, wf116,  (long)DFFd * Dd);
    cvt(stream, ffn_w2, wf216,  (long)Dd * DFFd);

    // ---- 1) self-attention projections (dual-pair fuses content+pos) ----
    gemm(stream, true, 1, tgt16, qpos16, wsa16 + 0 * WW, wsa16 + 1 * WW,
         sa_b + 0 * Dd, sa_b + 1 * Dd, q16, (int)MQ, Dd, Dd, Dd, Dd, Dd, Dd);
    gemm(stream, true, 1, tgt16, qpos16, wsa16 + 2 * WW, wsa16 + 3 * WW,
         sa_b + 2 * Dd, sa_b + 3 * Dd, k16, (int)MQ, Dd, Dd, Dd, Dd, Dd, Dd);
    gemm(stream, true, 1, tgt16, nullptr, wsa16 + 4 * WW, nullptr,
         sa_b + 4 * Dd, nullptr, v16, (int)MQ, Dd, Dd, Dd, Dd, Dd, Dd);

    // ---- 2) self scores: per (b,h)  [300x32] x [300x32]^T ----
    gemm(stream, true, 0, q16, nullptr, k16, nullptr, nullptr, nullptr, scS,
         NQd, NQd, hd, hd, LD, LD, SPAD,
         /*aOffB*/ Dd, /*aOffH*/ hd, /*bOffB*/ Dd, /*bOffH*/ hd,
         /*cOffB*/ (long)NHd * NQd * SPAD, /*cOffH*/ (long)NQd * SPAD,
         NHd, BBd * NHd);
    hipLaunchKernelGGL(softmax_kernel, dim3(BBd * NHd * NQd), dim3(256), 0,
                       stream, scS, atS16, NQd, SPAD, SPAD,
                       0.17677669529663687f);   // 1/sqrt(32)

    // ---- 3) attn @ V -> [L,B,D], then out_proj ----
    gemm(stream, false, 1, atS16, nullptr, v16, nullptr, nullptr, nullptr, ao16,
         NQd, hd, SPAD, NQd, SPAD, LD, LD,
         (long)NHd * NQd * SPAD, (long)NQd * SPAD, Dd, hd, Dd, hd,
         NHd, BBd * NHd);
    gemm(stream, true, 0, ao16, nullptr, wout16 + 0 * WW, nullptr,
         out_b + 0 * Dd, nullptr, aoF, (int)MQ, Dd, Dd, Dd, Dd, Dd, Dd);
    hipLaunchKernelGGL(add_ln_kernel, dim3((unsigned)MQ), dim3(Dd), 0, stream,
                       tgt, aoF, ln_g + 0 * Dd, ln_b + 0 * Dd, t1F, t1_16);

    // ---- 4) cross-attention projections ----
    gemm(stream, true, 1, t1_16, qpos16, wca16 + 0 * WW, wca16 + 4 * WW,
         ca_b + 0 * Dd, ca_b + 4 * Dd, qsum16, (int)MQ, Dd, Dd, Dd, Dd, Dd, Dd);
    gemm(stream, true, 1, qse16, nullptr, wca16 + 5 * WW, nullptr,
         ca_b + 5 * Dd, nullptr, qsep16, (int)MQ, Dd, Dd, Dd, Dd, Dd, Dd);
    gemm(stream, true, 1, mem16, pos16, wca16 + 1 * WW, wca16 + 3 * WW,
         ca_b + 1 * Dd, ca_b + 3 * Dd, ksum16, (int)MM, Dd, Dd, Dd, Dd, Dd, Dd);
    gemm(stream, true, 1, pos16, nullptr, wca16 + 3 * WW, nullptr,
         ca_b + 3 * Dd, nullptr, kpos16, (int)MM, Dd, Dd, Dd, Dd, Dd, Dd);
    gemm(stream, true, 1, mem16, nullptr, wca16 + 2 * WW, nullptr,
         ca_b + 2 * Dd, nullptr, vmem16, (int)MM, Dd, Dd, Dd, Dd, Dd, Dd);

    // ---- 5) cross scores: concat trick -> dual pair
    //      dot64(q||qse, k||kp) = dot32(qsum,ksum) + dot32(qsep,kpos) ----
    gemm(stream, true, 0, qsum16, qsep16, ksum16, kpos16, nullptr, nullptr, scC,
         NQd, HWd, hd, hd, LD, LD, HWd,
         Dd, hd, Dd, hd,
         (long)NHd * NQd * HWd, (long)NQd * HWd,
         NHd, BBd * NHd);
    hipLaunchKernelGGL(softmax_kernel, dim3(BBd * NHd * NQd), dim3(256), 0,
                       stream, scC, atC16, HWd, HWd, HWd, 0.125f); // 1/sqrt(64)

    // ---- 6) attn @ V_mem -> [L,B,D], out_proj, LN ----
    gemm(stream, false, 1, atC16, nullptr, vmem16, nullptr, nullptr, nullptr,
         ao2_16, NQd, hd, HWd, HWd, HWd, LD, LD,
         (long)NHd * NQd * HWd, (long)NQd * HWd, Dd, hd, Dd, hd,
         NHd, BBd * NHd);
    gemm(stream, true, 0, ao2_16, nullptr, wout16 + 1 * WW, nullptr,
         out_b + 1 * Dd, nullptr, ao2F, (int)MQ, Dd, Dd, Dd, Dd, Dd, Dd);
    hipLaunchKernelGGL(add_ln_kernel, dim3((unsigned)MQ), dim3(Dd), 0, stream,
                       t1F, ao2F, ln_g + 1 * Dd, ln_b + 1 * Dd, t2F, t2_16);

    // ---- 7) FFN: relu(t2 @ w1^T + b1) @ w2^T + b2, residual LN -> d_out ----
    gemm(stream, true, 2, t2_16, nullptr, wf116, nullptr, ffn_b1, nullptr, h16,
         (int)MQ, DFFd, Dd, Dd, Dd, Dd, DFFd);
    gemm(stream, true, 0, h16, nullptr, wf216, nullptr, ffn_b2, nullptr, ffnF,
         (int)MQ, Dd, DFFd, DFFd, DFFd, DFFd, Dd);
    hipLaunchKernelGGL(add_ln_kernel, dim3((unsigned)MQ), dim3(Dd), 0, stream,
                       t2F, ffnF, ln_g + 2 * Dd, ln_b + 2 * Dd,
                       (float*)d_out, (u16b*)nullptr);
}
